// MultiHeadHAAttentionLayer_3839700762657
// MI455X (gfx1250) — matrix operand order repaired and checked by reading, not compiled, not run on hardware
//
#include <hip/hip_runtime.h>

typedef __attribute__((ext_vector_type(16))) _Float16 v16h;
typedef __attribute__((ext_vector_type(8)))  float    v8f;

// ---------------- problem constants ----------------
static constexpr int kH  = 8;
static constexpr int kB  = 16;
static constexpr int kN  = 512;
static constexpr int kD  = 512;
static constexpr int kVD = 64;
static constexpr int kFFH = 2048;
static constexpr int kNP = 256;           // NPICK
static constexpr float kNORM = 0.125f;    // 1/sqrt(64)

// ---------------- generic batched WMMA GEMM ----------------
// C = epilogue( scale * A(MxK) @ B )   B is (K,N) row-major, or (N,K) if BT (=> A@B^T)
// batch: z = blockIdx.z ; z1 = z/batchDiv ; z2 = z%batchDiv ; ptr += z1*s?1 + z2*s?2
// Tile: 64x64x32, 128 threads (4 waves), wave w owns rows [16w,16w+16), 4 col-tiles.
#define TM 64
#define TN 64
#define TK 32

template<bool BT, bool HAS_BIAS, bool RELU, bool HAS_RES, bool HAS_ALPHA>
__global__ void __launch_bounds__(128)
gemm_wmma(const float* __restrict__ Ab, const float* __restrict__ Bb,
          const float* __restrict__ ResB, const float* __restrict__ bias,
          const float* __restrict__ alphaPtr, float* __restrict__ Cb,
          int M, int Ncols, int K, int lda, int ldb, int ldc,
          int batchDiv,
          long sA1, long sA2, long sB1, long sB2, long sC1, long sC2,
          float scale)
{
    __shared__ _Float16 As[TM][TK + 2];
    __shared__ _Float16 Bs[TN][TK + 2];

    const int z  = blockIdx.z;
    const int z1 = z / batchDiv;
    const int z2 = z % batchDiv;
    const float* A = Ab + z1 * sA1 + z2 * sA2;
    const float* Bm = Bb + z1 * sB1 + z2 * sB2;
    float*       C = Cb + z1 * sC1 + z2 * sC2;
    const float* Res = HAS_RES ? (ResB + z1 * sC1 + z2 * sC2) : nullptr;

    const int m0 = blockIdx.y * TM;
    const int n0 = blockIdx.x * TN;
    const int t    = threadIdx.x;
    const int lane = t & 31;
    const int wave = t >> 5;
    const int r    = lane & 15;
    const int hi   = lane >> 4;

    v8f acc[4] = {};

    for (int k0 = 0; k0 < K; k0 += TK) {
        // cooperative A tile: 64x32 (f32 -> f16)
        #pragma unroll
        for (int i = 0; i < 16; ++i) {
            int idx = i * 128 + t;
            int row = idx >> 5;
            int kk  = idx & 31;
            As[row][kk] = (_Float16)A[(long)(m0 + row) * lda + (k0 + kk)];
        }
        // cooperative B tile stored [n][k]
        #pragma unroll
        for (int i = 0; i < 16; ++i) {
            int idx = i * 128 + t;
            if (BT) {
                int col = idx >> 5;
                int kk  = idx & 31;
                Bs[col][kk] = (_Float16)Bm[(long)(n0 + col) * ldb + (k0 + kk)];
            } else {
                int kk  = idx >> 6;
                int col = idx & 63;
                Bs[col][kk] = (_Float16)Bm[(long)(k0 + kk) * ldb + (n0 + col)];
            }
        }
        __syncthreads();

        // A fragment: lane r holds row (16*wave + r); k pattern per ISA 16-bit A layout
        v16h af;
        const int arow = wave * 16 + r;
        #pragma unroll
        for (int v = 0; v < 8; ++v) {
            int kb = (v >> 2) * 16 + hi * 8 + (v & 3) * 2;
            af[2 * v]     = As[arow][kb];
            af[2 * v + 1] = As[arow][kb + 1];
        }
        #pragma unroll
        for (int ct = 0; ct < 4; ++ct) {
            v16h bf;
            const int bcol = ct * 16 + r;
            #pragma unroll
            for (int v = 0; v < 8; ++v) {
                int kb = (v >> 2) * 16 + hi * 8 + (v & 3) * 2;
                bf[2 * v]     = Bs[bcol][kb];
                bf[2 * v + 1] = Bs[bcol][kb + 1];
            }
            acc[ct] = __builtin_amdgcn_wmma_f32_16x16x32_f16(
                false, af, false, bf, (short)0, acc[ct], false, false);
        }
        __syncthreads();
    }

    float alpha = 1.0f;
    if (HAS_ALPHA) alpha = *alphaPtr;

    #pragma unroll
    for (int ct = 0; ct < 4; ++ct) {
        #pragma unroll
        for (int v = 0; v < 8; ++v) {
            int gm = m0 + wave * 16 + v + 8 * hi;
            int gn = n0 + ct * 16 + r;
            float val = acc[ct][v] * scale;
            if (HAS_BIAS) val += bias[gn];
            if (RELU)     val = val > 0.0f ? val : 0.0f;
            if (HAS_ALPHA) val *= alpha;
            long cidx = (long)gm * ldc + gn;
            if (HAS_RES) val += Res[cidx];
            C[cidx] = val;
        }
    }
}

// ---------------- c_pd / c_dp row-dots ----------------
// scol[z*N + n] = NORM * dot(QA[z,n,:], K[z,(n+256)%512,:])
__global__ void ha_coldot(const float* __restrict__ QA, const float* __restrict__ Km,
                          float* __restrict__ scol)
{
    long i = (long)blockIdx.x * blockDim.x + threadIdx.x;   // over 128*512
    long z = i >> 9;
    int  n = (int)(i & 511);
    int  partner = (n + kNP) & (kN - 1);
    const float* qa = QA + z * ((long)kN * kVD) + (long)n * kVD;
    const float* kk = Km + z * ((long)kN * kVD) + (long)partner * kVD;
    float acc = 0.0f;
    #pragma unroll
    for (int j = 0; j < kVD; ++j) acc += qa[j] * kk[j];
    scol[i] = acc * kNORM;
}

// ---------------- ha softmax + fold (in place into Sm) ----------------
// Row n of batch z: finite scores = main[512] + same[256] + opp[256] + col(1).
// After softmax, fold everything onto a 512-wide weight vector over V rows.
__global__ void __launch_bounds__(256)
ha_softmax_fold(float* __restrict__ Sm, const float* __restrict__ Ss,
                const float* __restrict__ So, const float* __restrict__ Sc)
{
    const int n = blockIdx.x;           // 0..511
    const int z = blockIdx.y;           // h*B + b
    float*       pm = Sm + (long)z * kN * kN + (long)n * kN;
    const float* ps = Ss + (long)z * kN * kNP + (long)n * kNP;
    const float* po = So + (long)z * kN * kNP + (long)n * kNP;
    const float  c  = Sc[(long)z * kN + n];
    const int t = threadIdx.x;          // 0..255
    const bool pick = (n < kNP);
    const int partner = (n + kNP) & (kN - 1);

    float m0 = pm[t], m1 = pm[t + kNP], sv = ps[t], ov = po[t];

    __shared__ float red[256];
    float lmax = fmaxf(fmaxf(m0, m1), fmaxf(sv, ov));
    if (t == 0) lmax = fmaxf(lmax, c);
    red[t] = lmax; __syncthreads();
    for (int s = 128; s > 0; s >>= 1) { if (t < s) red[t] = fmaxf(red[t], red[t + s]); __syncthreads(); }
    const float mx = red[0];
    __syncthreads();

    float e0 = __expf(m0 - mx), e1 = __expf(m1 - mx);
    float es = __expf(sv - mx), eo = __expf(ov - mx);
    float ec = __expf(c  - mx);
    float lsum = e0 + e1 + es + eo;
    if (t == 0) lsum += ec;
    red[t] = lsum; __syncthreads();
    for (int s = 128; s > 0; s >>= 1) { if (t < s) red[t] += red[t + s]; __syncthreads(); }
    const float inv = 1.0f / red[0];

    float w0, w1;
    if (pick) { w0 = e0 + es; w1 = e1 + eo; }   // pap -> V[0:256], pad -> V[256:512]
    else      { w0 = e0 + eo; w1 = e1 + es; }   // dap -> V[0:256], dad -> V[256:512]
    if (t == (partner & (kNP - 1))) {
        if (pick) w1 += ec;   // col_pd -> V[n+256]
        else      w0 += ec;   // col_dp -> V[n-256]
    }
    pm[t]       = w0 * inv;
    pm[t + kNP] = w1 * inv;
}

// ---------------- plain softmax over rows of width 512 ----------------
__global__ void __launch_bounds__(256)
softmax512(float* __restrict__ S)
{
    float* p = S + (long)blockIdx.x * 512;
    const int t = threadIdx.x;
    float a = p[t], b = p[t + 256];
    __shared__ float red[256];
    red[t] = fmaxf(a, b); __syncthreads();
    for (int s = 128; s > 0; s >>= 1) { if (t < s) red[t] = fmaxf(red[t], red[t + s]); __syncthreads(); }
    const float mx = red[0];
    __syncthreads();
    float ea = __expf(a - mx), eb = __expf(b - mx);
    red[t] = ea + eb; __syncthreads();
    for (int s = 128; s > 0; s >>= 1) { if (t < s) red[t] += red[t + s]; __syncthreads(); }
    const float inv = 1.0f / red[0];
    p[t] = ea * inv;
    p[t + 256] = eb * inv;
}

// ---------------- host-side GEMM dispatch ----------------
// mode 0: plain   mode 1: bias + *alpha + residual   mode 2: bias + relu
static void launch_gemm(hipStream_t st, int mode, bool bt,
                        const float* A, const float* B, const float* Res,
                        const float* bias, const float* alpha, float* C,
                        int M, int N, int K, int lda, int ldb, int ldc,
                        int batches, int batchDiv,
                        long sA1, long sA2, long sB1, long sB2, long sC1, long sC2,
                        float scale)
{
    dim3 grid(N / TN, M / TM, batches);
    dim3 block(128);
    if (bt) {
        gemm_wmma<true, false, false, false, false><<<grid, block, 0, st>>>(
            A, B, Res, bias, alpha, C, M, N, K, lda, ldb, ldc,
            batchDiv, sA1, sA2, sB1, sB2, sC1, sC2, scale);
    } else if (mode == 0) {
        gemm_wmma<false, false, false, false, false><<<grid, block, 0, st>>>(
            A, B, Res, bias, alpha, C, M, N, K, lda, ldb, ldc,
            batchDiv, sA1, sA2, sB1, sB2, sC1, sC2, scale);
    } else if (mode == 1) {
        gemm_wmma<false, true, false, true, true><<<grid, block, 0, st>>>(
            A, B, Res, bias, alpha, C, M, N, K, lda, ldb, ldc,
            batchDiv, sA1, sA2, sB1, sB2, sC1, sC2, scale);
    } else {
        gemm_wmma<false, true, true, false, false><<<grid, block, 0, st>>>(
            A, B, Res, bias, alpha, C, M, N, K, lda, ldb, ldc,
            batchDiv, sA1, sA2, sB1, sB2, sC1, sC2, scale);
    }
}

extern "C" void kernel_launch(void* const* d_in, const int* in_sizes, int n_in,
                              void* d_out, int out_size, void* d_ws, size_t ws_size,
                              hipStream_t stream)
{
    (void)in_sizes; (void)n_in; (void)out_size; (void)ws_size;

    const float* x        = (const float*)d_in[0];
    const float* Wqry     = (const float*)d_in[1];
    const float* Wkey     = (const float*)d_in[2];
    const float* Wval     = (const float*)d_in[3];
    const float* W1q      = (const float*)d_in[4];
    const float* W2q      = (const float*)d_in[5];
    const float* W3q      = (const float*)d_in[6];
    const float* W4q      = (const float*)d_in[7];
    const float* W5q      = (const float*)d_in[8];
    const float* W6q      = (const float*)d_in[9];
    const float* Wq_w     = (const float*)d_in[10];
    const float* Wk_w     = (const float*)d_in[11];
    const float* Wv_w     = (const float*)d_in[12];
    const float* Wq2p_w   = (const float*)d_in[13];
    const float* Wq2d_w   = (const float*)d_in[14];
    const float* Wk2_w    = (const float*)d_in[15];
    const float* Wv2_w    = (const float*)d_in[16];
    const float* comb_w   = (const float*)d_in[17];
    const float* comb_b   = (const float*)d_in[18];
    const float* comb2_w  = (const float*)d_in[19];
    const float* comb2_b  = (const float*)d_in[20];
    const float* comb3_w  = (const float*)d_in[21];
    const float* comb3_b  = (const float*)d_in[22];
    const float* alpha1   = (const float*)d_in[23];
    const float* alpha2   = (const float*)d_in[24];
    const float* alpha3   = (const float*)d_in[25];
    const float* ff1_w1   = (const float*)d_in[26];
    const float* ff1_b1   = (const float*)d_in[27];
    const float* ff1_w2   = (const float*)d_in[28];
    const float* ff1_b2   = (const float*)d_in[29];
    const float* ffa1     = (const float*)d_in[30];
    const float* ff2_w1   = (const float*)d_in[31];
    const float* ff2_b1   = (const float*)d_in[32];
    const float* ff2_w2   = (const float*)d_in[33];
    const float* ff2_b2   = (const float*)d_in[34];
    const float* ffa2     = (const float*)d_in[35];
    const float* ff3_w1   = (const float*)d_in[36];
    const float* ff3_b1   = (const float*)d_in[37];
    const float* ff3_w2   = (const float*)d_in[38];
    const float* ff3_b2   = (const float*)d_in[39];
    const float* ffa3     = (const float*)d_in[40];

    const long BN   = (long)kB * kN;          // 8192
    const long NVD  = (long)kN * kVD;         // 32768
    const long PROJ = (long)kH * kB * NVD;    // 4,194,304
    const long HB   = (long)kH * kB;          // 128

    float* ws = (float*)d_ws;
    float* Qm = ws;                  // (H,B,N,VD)
    float* Km = Qm + PROJ;
    float* Vm = Km + PROJ;
    float* QA = Vm + PROJ;           // W1 (pick) / W4 (dlv)
    float* QB = QA + PROJ;           // W2 (pick) / W5 (dlv)  -> same-half block
    float* QC = QB + PROJ;           // W3 (pick) / W6 (dlv)  -> opposite-half block
    float* Sm = QC + PROJ;           // (HB, N, N) main scores / folded weights; reused as MHA scores
    float* Ss = Sm + HB * (long)kN * kN;
    float* So = Ss + HB * (long)kN * kNP;
    float* Sc = So + HB * (long)kN * kNP;
    float* Ha = Sc + HB * (long)kN;  // (B,N,512)
    float* OutA = Ha + BN * 512;
    float* OutB = OutA + BN * 512;
    float* Tbuf = OutB + BN * 512;   // (B*N, FFH)
    float* Qb = Qm; float* Kb = Km; float* Vb = Vm;   // reuse for MHA q/k/v
    float* Msc = Sm;                                  // reuse for MHA scores

    const long WH = (long)kD * kVD;  // 32768, per-head weight stride

    // ---- ha_encoder projections ----
    // Q/K/V: batch over h
    launch_gemm(stream, 0, false, x, Wqry, nullptr, nullptr, nullptr, Qm,
                (int)BN, kVD, kD, kD, kVD, kVD, kH, 1, 0, 0, WH, 0, BN * kVD, 0, 1.0f);
    launch_gemm(stream, 0, false, x, Wkey, nullptr, nullptr, nullptr, Km,
                (int)BN, kVD, kD, kD, kVD, kVD, kH, 1, 0, 0, WH, 0, BN * kVD, 0, 1.0f);
    launch_gemm(stream, 0, false, x, Wval, nullptr, nullptr, nullptr, Vm,
                (int)BN, kVD, kD, kD, kVD, kVD, kH, 1, 0, 0, WH, 0, BN * kVD, 0, 1.0f);
    // QA/QB/QC: pick rows use W1/W2/W3, dlv rows use W4/W5/W6; batch z = h*B + b
    const float* Wp[3] = { W1q, W2q, W3q };
    const float* Wd[3] = { W4q, W5q, W6q };
    float* Qx[3] = { QA, QB, QC };
    for (int i = 0; i < 3; ++i) {
        launch_gemm(stream, 0, false, x, Wp[i], nullptr, nullptr, nullptr, Qx[i],
                    kNP, kVD, kD, kD, kVD, kVD, (int)HB, kB,
                    0, (long)kN * kD, WH, 0, (long)kB * NVD, NVD, 1.0f);
        launch_gemm(stream, 0, false, x + (long)kNP * kD, Wd[i], nullptr, nullptr, nullptr,
                    Qx[i] + (long)kNP * kVD,
                    kNP, kVD, kD, kD, kVD, kVD, (int)HB, kB,
                    0, (long)kN * kD, WH, 0, (long)kB * NVD, NVD, 1.0f);
    }

    // ---- ha scores ----
    // main: Q @ K^T per (h,b)
    launch_gemm(stream, 0, true, Qm, Km, nullptr, nullptr, nullptr, Sm,
                kN, kN, kVD, kVD, kVD, kN, (int)HB, 1,
                NVD, 0, NVD, 0, (long)kN * kN, 0, kNORM);
    // same-half block (pap / dad)
    launch_gemm(stream, 0, true, QB, Km, nullptr, nullptr, nullptr, Ss,
                kNP, kNP, kVD, kVD, kVD, kNP, (int)HB, 1,
                NVD, 0, NVD, 0, (long)kN * kNP, 0, kNORM);
    launch_gemm(stream, 0, true, QB + (long)kNP * kVD, Km + (long)kNP * kVD, nullptr, nullptr, nullptr,
                Ss + (long)kNP * kNP,
                kNP, kNP, kVD, kVD, kVD, kNP, (int)HB, 1,
                NVD, 0, NVD, 0, (long)kN * kNP, 0, kNORM);
    // opposite-half block (pad / dap)
    launch_gemm(stream, 0, true, QC, Km + (long)kNP * kVD, nullptr, nullptr, nullptr, So,
                kNP, kNP, kVD, kVD, kVD, kNP, (int)HB, 1,
                NVD, 0, NVD, 0, (long)kN * kNP, 0, kNORM);
    launch_gemm(stream, 0, true, QC + (long)kNP * kVD, Km, nullptr, nullptr, nullptr,
                So + (long)kNP * kNP,
                kNP, kNP, kVD, kVD, kVD, kNP, (int)HB, 1,
                NVD, 0, NVD, 0, (long)kN * kNP, 0, kNORM);
    // col dots
    ha_coldot<<<(int)(HB * kN / 256), 256, 0, stream>>>(QA, Km, Sc);

    // ---- softmax + fold (in place into Sm) ----
    ha_softmax_fold<<<dim3(kN, (unsigned)HB), 256, 0, stream>>>(Sm, Ss, So, Sc);

    // ---- AV with transpose-to-(B,N,H*VD) epilogue ----
    launch_gemm(stream, 0, false, Sm, Vm, nullptr, nullptr, nullptr, Ha,
                kN, kVD, kN, kN, kVD, 512, (int)HB, kB,
                (long)kB * kN * kN, (long)kN * kN,
                (long)kB * NVD, NVD,
                64, (long)kN * 512, 1.0f);

    // ---- out = x + (ha @ comb3 + b3)*alpha3 ; ff3 ----
    launch_gemm(stream, 1, false, Ha, comb3_w, x, comb3_b, alpha3, OutA,
                (int)BN, kD, 512, 512, kD, kD, 1, 1, 0, 0, 0, 0, 0, 0, 1.0f);
    launch_gemm(stream, 2, false, OutA, ff3_w1, nullptr, ff3_b1, nullptr, Tbuf,
                (int)BN, kFFH, kD, kD, kFFH, kFFH, 1, 1, 0, 0, 0, 0, 0, 0, 1.0f);
    launch_gemm(stream, 1, false, Tbuf, ff3_w2, OutA, ff3_b2, ffa3, OutB,
                (int)BN, kD, kFFH, kFFH, kD, kD, 1, 1, 0, 0, 0, 0, 0, 0, 1.0f);

    // ---- MHA 1 ----
    launch_gemm(stream, 0, false, OutB, Wq_w, nullptr, nullptr, nullptr, Qb,
                (int)BN, 512, kD, kD, 512, 512, 1, 1, 0, 0, 0, 0, 0, 0, 1.0f);
    launch_gemm(stream, 0, false, OutB, Wk_w, nullptr, nullptr, nullptr, Kb,
                (int)BN, 512, kD, kD, 512, 512, 1, 1, 0, 0, 0, 0, 0, 0, 1.0f);
    launch_gemm(stream, 0, false, OutB, Wv_w, nullptr, nullptr, nullptr, Vb,
                (int)BN, 512, kD, kD, 512, 512, 1, 1, 0, 0, 0, 0, 0, 0, 1.0f);
    for (int rep = 0; rep < 2; ++rep) {
        // scores: z = b*H + h
        launch_gemm(stream, 0, true, Qb, Kb, nullptr, nullptr, nullptr, Msc,
                    kN, kN, kVD, 512, 512, kN, (int)HB, kH,
                    (long)kN * 512, 64, (long)kN * 512, 64,
                    (long)kH * kN * kN, (long)kN * kN, 0.125f);
        softmax512<<<(int)(HB * kN), 256, 0, stream>>>(Msc);
        // AV folded back to (B,N,512)
        launch_gemm(stream, 0, false, Msc, Vb, nullptr, nullptr, nullptr, Ha,
                    kN, kVD, kN, kN, 512, 512, (int)HB, kH,
                    (long)kH * kN * kN, (long)kN * kN,
                    (long)kN * 512, 64,
                    (long)kN * 512, 64, 1.0f);
        if (rep == 0) {
            launch_gemm(stream, 1, false, Ha, comb_w, OutB, comb_b, alpha1, OutA,
                        (int)BN, kD, 512, 512, kD, kD, 1, 1, 0, 0, 0, 0, 0, 0, 1.0f);
            launch_gemm(stream, 2, false, OutA, ff1_w1, nullptr, ff1_b1, nullptr, Tbuf,
                        (int)BN, kFFH, kD, kD, kFFH, kFFH, 1, 1, 0, 0, 0, 0, 0, 0, 1.0f);
            launch_gemm(stream, 1, false, Tbuf, ff1_w2, OutA, ff1_b2, ffa1, OutB,
                        (int)BN, kD, kFFH, kFFH, kD, kD, 1, 1, 0, 0, 0, 0, 0, 0, 1.0f);
            // MHA2 projections: q2 pick/dlv, then k2/v2
            launch_gemm(stream, 0, false, OutB, Wq2p_w, nullptr, nullptr, nullptr, Qb,
                        kNP, 512, kD, kD, 512, 512, kB, 1,
                        (long)kN * 512, 0, 0, 0, (long)kN * 512, 0, 1.0f);
            launch_gemm(stream, 0, false, OutB + (long)kNP * kD, Wq2d_w, nullptr, nullptr, nullptr,
                        Qb + (long)kNP * 512,
                        kNP, 512, kD, kD, 512, 512, kB, 1,
                        (long)kN * 512, 0, 0, 0, (long)kN * 512, 0, 1.0f);
            launch_gemm(stream, 0, false, OutB, Wk2_w, nullptr, nullptr, nullptr, Kb,
                        (int)BN, 512, kD, kD, 512, 512, 1, 1, 0, 0, 0, 0, 0, 0, 1.0f);
            launch_gemm(stream, 0, false, OutB, Wv2_w, nullptr, nullptr, nullptr, Vb,
                        (int)BN, 512, kD, kD, 512, 512, 1, 1, 0, 0, 0, 0, 0, 0, 1.0f);
        } else {
            launch_gemm(stream, 1, false, Ha, comb2_w, OutB, comb2_b, alpha2, OutA,
                        (int)BN, kD, 512, 512, kD, kD, 1, 1, 0, 0, 0, 0, 0, 0, 1.0f);
            launch_gemm(stream, 2, false, OutA, ff2_w1, nullptr, ff2_b1, nullptr, Tbuf,
                        (int)BN, kFFH, kD, kD, kFFH, kFFH, 1, 1, 0, 0, 0, 0, 0, 0, 1.0f);
            launch_gemm(stream, 1, false, Tbuf, ff2_w2, OutA, ff2_b2, ffa2, (float*)d_out,
                        (int)BN, kD, kFFH, kFFH, kD, kD, 1, 1, 0, 0, 0, 0, 0, 0, 1.0f);
        }
    }
}
